// KANFFN_5909874999577
// MI455X (gfx1250) — matrix operand order repaired and checked by reading, not compiled, not run on hardware
//
#include <hip/hip_runtime.h>
#include <hip/hip_fp16.h>

typedef __attribute__((ext_vector_type(16))) _Float16 v16h;
typedef __attribute__((ext_vector_type(2)))  __fp16   fp16x2;  // cvt_pkrtz return type
typedef __attribute__((ext_vector_type(8)))  float    v8f;

#define NJ 9            // 1 silu plane + 8 spline basis planes
#define BM 128
#define CF 32           // input features per K-chunk (one WMMA k-step per j-plane)
#define A_STRIDE 32     // f16 elems per A-tile row
#define W_STRIDE 296    // f16 elems per W-tile row (288 data + 8 pad for banks)
#define A_ELEMS (NJ * BM * A_STRIDE)     // 36864 f16 = 73.7 KB

// f32 -> f16 bits, RNE (one-time weight prep).
__device__ __forceinline__ unsigned short f2h(float f) {
  union { _Float16 h; unsigned short u; } cv;
  cv.h = (_Float16)f;
  return cv.u;
}

// Two f32 -> packed f16 pair in one v_cvt_pk_rtz_f16_f32.
__device__ __forceinline__ unsigned int pack_f16x2(float lo, float hi) {
  union { fp16x2 h; unsigned int u; } cv;
  cv.h = __builtin_amdgcn_cvt_pkrtz(lo, hi);
  return cv.u;
}

// Evaluate silu(x) and all 8 cubic B-spline basis values on the uniform grid
// (knots t_r = -2.2 + 0.4 r, r=0..11) via closed-form local polynomials:
// s=(x-t0)/h, c=floor(s), u=s-c; only bases g=c-3..c are nonzero with the
// standard uniform cubic weights. Matches the Cox-de Boor recursion exactly
// (edges: missing bases are zero; x outside the grid -> all-zero one-hot).
__device__ __forceinline__ void kan_eval9(float x, float out[NJ]) {
  out[0] = x * (1.0f / (1.0f + __expf(-x)));          // silu base term

  const float s = (x + 2.2f) * 2.5f;                  // knot-space coordinate
  const float c = floorf(s);
  const float u = s - c;
  const float k6 = 1.0f / 6.0f;
  const float omu = 1.0f - u;
  const float u2 = u * u;
  const float n3 = u2 * u * k6;                       // g == c
  const float n0 = omu * omu * omu * k6;              // g == c-3
  float a = __builtin_fmaf(-3.0f, u, 3.0f);           // n2 = (-3u^3+3u^2+3u+1)/6
  a = __builtin_fmaf(a, u, 3.0f);
  const float n2 = __builtin_fmaf(a, u, 1.0f) * k6;   // g == c-1
  float bqa = __builtin_fmaf(3.0f, u, -6.0f);         // n1 = (3u^3-6u^2+4)/6
  bqa = bqa * u;
  const float n1 = __builtin_fmaf(bqa, u, 4.0f) * k6; // g == c-2

  float e[11];                                        // one-hot interval vector
#pragma unroll
  for (int m = 0; m < 11; ++m) e[m] = (c == (float)m) ? 1.0f : 0.0f;
#pragma unroll
  for (int g = 0; g < 8; ++g) {
    float v = n3 * e[g];
    v = __builtin_fmaf(n2, e[g + 1], v);
    v = __builtin_fmaf(n1, e[g + 2], v);
    v = __builtin_fmaf(n0, e[g + 3], v);
    out[1 + g] = v;
  }
}

// Pack W^T[o][j*Kin + i] = f16( j==0 ? scale_base[i,o] : coef[i,o,j-1]*scale_sp[i,o] )
// Row-major (Ncols x 9*Kin) so B-fragments are contiguous in K.
__global__ __launch_bounds__(256) void kan_prep_w(
    const float* __restrict__ coef, const float* __restrict__ sbase,
    const float* __restrict__ ssp, unsigned short* __restrict__ Wt,
    int Kin, int Ncols)
{
  int t = blockIdx.x * blockDim.x + threadIdx.x;
  if (t >= Kin * Ncols) return;
  int i = t % Kin;          // fast index -> coalesced f16 stores
  int o = t / Kin;
  size_t io = (size_t)i * Ncols + o;
  float sb = sbase[io];
  float sp = ssp[io];
  const float* cf = coef + io * 8;
  unsigned short* wrow = Wt + (size_t)o * ((size_t)NJ * Kin);
  wrow[i] = f2h(sb);
#pragma unroll
  for (int g = 0; g < 8; ++g)
    wrow[(size_t)(g + 1) * Kin + i] = f2h(cf[g] * sp);
}

// Fused KAN layer: Y(Nrows x Ncols) = A_exp(X) @ W^T, basis expansion on the fly.
// Block = BM x TBN output tile; each wave computes 64x32 via 4x2 WMMA f16 tiles.
// TBN=256/512thr (layer 1, matrix-bound) or TBN=128/256thr (layer 2, full grid).
template<int TBN, int TTHREADS>
__global__ __launch_bounds__(TTHREADS) void kan_fused_gemm(
    const float* __restrict__ X, const unsigned short* __restrict__ Wt,
    float* __restrict__ Y, int Kin, int Ncols)
{
  constexpr int NWAVE = TTHREADS / 32;
  constexpr int WCOLS = TBN / 32;            // column wave-groups
  constexpr int WROWS = NWAVE / WCOLS;       // must be 2 (BM/64)
  static_assert(WROWS == 2, "wave grid must be 2 x (TBN/32)");
  constexpr int EPT    = (BM * CF) / TTHREADS;   // A elements per thread (8 or 16)
  constexpr int GROUPS = EPT / 4;
  constexpr int TPR    = TTHREADS / BM;          // threads per A row (4 or 2)
  constexpr int UNITS  = (TBN * 36) / TTHREADS;  // 16B W units per thread (18)

  extern __shared__ unsigned short lds[];
  unsigned short* Alds = lds;             // [NJ][BM][A_STRIDE] f16
  unsigned short* Wlds = lds + A_ELEMS;   // [TBN][W_STRIDE]    f16

  const int tid  = threadIdx.x;
  const int lane = tid & 31;
  const int wave = tid >> 5;
  const int wrow = wave / WCOLS;           // 0..1  (row group of 64)
  const int wcol = wave % WCOLS;           // 0..WCOLS-1 (col group of 32)
  const int m15  = lane & 15;
  const int hi   = lane >> 4;

  const int tileN0 = blockIdx.x * TBN;
  const int tileM0 = blockIdx.y * BM;
  const size_t Kexp = (size_t)NJ * Kin;

  const int arow = tid / TPR;              // A-staging: row in tile
  const int af0  = (tid % TPR) * EPT;      // A-staging: first feature
  const float* xrow = X + (size_t)(tileM0 + arow) * Kin;

  v8f acc[4][2];
  const v8f vzero = {0.f, 0.f, 0.f, 0.f, 0.f, 0.f, 0.f, 0.f};
#pragma unroll
  for (int mt = 0; mt < 4; ++mt)
#pragma unroll
    for (int nt = 0; nt < 2; ++nt) acc[mt][nt] = vzero;

#pragma unroll 1
  for (int i0 = 0; i0 < Kin; i0 += CF) {
    // ---- stage expanded-A tile: silu + 8 basis planes, f32 -> f16 -> LDS ----
    {
      const float4* xp = (const float4*)(xrow + i0 + af0);
      float4 xq[GROUPS];
#pragma unroll
      for (int g = 0; g < GROUPS; ++g) xq[g] = xp[g];
#pragma unroll
      for (int g = 0; g < GROUPS; ++g) {
        float ov[4][NJ];
        kan_eval9(xq[g].x, ov[0]);
        kan_eval9(xq[g].y, ov[1]);
        kan_eval9(xq[g].z, ov[2]);
        kan_eval9(xq[g].w, ov[3]);
#pragma unroll
        for (int j = 0; j < NJ; ++j) {
          uint2 p;
          p.x = pack_f16x2(ov[0][j], ov[1][j]);
          p.y = pack_f16x2(ov[2][j], ov[3][j]);
          *(uint2*)&Alds[j * (BM * A_STRIDE) + arow * A_STRIDE + af0 + g * 4] = p;
        }
      }
    }
    // ---- stage W^T tile: TBN rows x (NJ*CF) contiguous-K f16, b128 copies ----
    {
#pragma unroll
      for (int s = 0; s < UNITS; ++s) {
        int u   = tid + s * TTHREADS;     // 16B-units, 36 per output column
        int o   = u / 36;
        int rem = u - o * 36;
        int j   = rem >> 2;
        int q   = rem & 3;
        const uint4* src = (const uint4*)(Wt + (size_t)(tileN0 + o) * Kexp +
                                          (size_t)j * Kin + i0 + q * 8);
        *(uint4*)&Wlds[o * W_STRIDE + j * CF + q * 8] = *src;
      }
    }
    __syncthreads();

    // ---- 9 WMMA k-steps (one per expanded plane) ----
    for (int j = 0; j < NJ; ++j) {
      union { uint4 q[2]; v16h v; } ua[4], ub[2];
#pragma unroll
      for (int mt = 0; mt < 4; ++mt) {
        // A 16x32 f16 frag: row = m15, runs K = hi*8.. and 16+hi*8..
        const unsigned short* pa =
            &Alds[j * (BM * A_STRIDE) + (wrow * 64 + mt * 16 + m15) * A_STRIDE + hi * 8];
        ua[mt].q[0] = *(const uint4*)(pa);
        ua[mt].q[1] = *(const uint4*)(pa + 16);
      }
#pragma unroll
      for (int nt = 0; nt < 2; ++nt) {
        // B 32x16 f16 frag: col = m15, runs K = hi*8.. and 16+hi*8..
        const unsigned short* pb =
            &Wlds[(wcol * 32 + nt * 16 + m15) * W_STRIDE + j * CF + hi * 8];
        ub[nt].q[0] = *(const uint4*)(pb);
        ub[nt].q[1] = *(const uint4*)(pb + 16);
      }
#pragma unroll
      for (int mt = 0; mt < 4; ++mt)
#pragma unroll
        for (int nt = 0; nt < 2; ++nt)
          acc[mt][nt] = __builtin_amdgcn_wmma_f32_16x16x32_f16(
              false, ua[mt].v, false, ub[nt].v, (short)0, acc[mt][nt], false, false);
    }
    __syncthreads();
  }

  // ---- write f32 result (C frag: VGPR v -> M = hi*8+v, N = m15) ----
#pragma unroll
  for (int mt = 0; mt < 4; ++mt) {
#pragma unroll
    for (int nt = 0; nt < 2; ++nt) {
      const int col = tileN0 + wcol * 32 + nt * 16 + m15;
#pragma unroll
      for (int v = 0; v < 8; ++v) {
        const int row = tileM0 + wrow * 64 + mt * 16 + hi * 8 + v;
        Y[(size_t)row * Ncols + col] = acc[mt][nt][v];
      }
    }
  }
}

extern "C" void kernel_launch(void* const* d_in, const int* in_sizes, int n_in,
                              void* d_out, int out_size, void* d_ws, size_t ws_size,
                              hipStream_t stream) {
  const float* x     = (const float*)d_in[0];
  const float* coef1 = (const float*)d_in[1];
  const float* sb1   = (const float*)d_in[2];
  const float* sp1   = (const float*)d_in[3];
  const float* coef2 = (const float*)d_in[4];
  const float* sb2   = (const float*)d_in[5];
  const float* sp2   = (const float*)d_in[6];
  float* out = (float*)d_out;

  const int HID = 512, FFN = 2048;
  const int nrows = in_sizes[0] / HID;   // (b*s) = 8192

  unsigned char* ws = (unsigned char*)d_ws;
  unsigned short* W1t = (unsigned short*)ws;                      // FFN x 9*HID f16
  size_t w1_bytes = (size_t)FFN * NJ * HID * 2;                   // 18.9 MB
  unsigned short* W2t = (unsigned short*)(ws + w1_bytes);         // HID x 9*FFN f16
  size_t w2_bytes = (size_t)HID * NJ * FFN * 2;                   // 18.9 MB
  float* h = (float*)(ws + w1_bytes + w2_bytes);                  // nrows x FFN f32 (67 MB)

  kan_prep_w<<<(HID * FFN + 255) / 256, 256, 0, stream>>>(coef1, sb1, sp1, W1t, HID, FFN);
  kan_prep_w<<<(FFN * HID + 255) / 256, 256, 0, stream>>>(coef2, sb2, sp2, W2t, FFN, HID);

  // Layer 1: BN=256, 512 threads -> eval VALU halves, matrix-bound; 512 WGs.
  {
    constexpr int TBN = 256, TTH = 512;
    size_t ldsb = (size_t)(A_ELEMS + TBN * W_STRIDE) * 2;   // 225.3 KB -> 1 WG/WGP
    dim3 g1(FFN / TBN, nrows / BM);
    kan_fused_gemm<TBN, TTH><<<g1, TTH, ldsb, stream>>>(x, W1t, h, HID, FFN);
  }
  // Layer 2: BN=128, 256 threads -> 256 WGs keep all WGPs busy.
  {
    constexpr int TBN = 128, TTH = 256;
    size_t ldsb = (size_t)(A_ELEMS + TBN * W_STRIDE) * 2;   // 149.5 KB -> 2 WG/WGP
    dim3 g2(HID / TBN, nrows / BM);
    kan_fused_gemm<TBN, TTH><<<g2, TTH, ldsb, stream>>>(h, W2t, out, FFN, HID);
  }
}
